// PatternEncode_16638703305225
// MI455X (gfx1250) — matrix operand order repaired and checked by reading, not compiled, not run on hardware
//
#include <hip/hip_runtime.h>
#include <math.h>

typedef __attribute__((ext_vector_type(16))) _Float16 v16h;
typedef __attribute__((ext_vector_type(8)))  float    v8f;
typedef __attribute__((ext_vector_type(4)))  unsigned int u32x4;
typedef __attribute__((ext_vector_type(8)))  int      i32x8;
typedef __attribute__((ext_vector_type(4)))  int      i32x4;

union V16 { v16h v; _Float16 e[16]; };

#if defined(__has_builtin)
#if __has_builtin(__builtin_amdgcn_tensor_load_to_lds) && __has_builtin(__builtin_amdgcn_s_wait_tensorcnt)
#define USE_TDM 1
#endif
#endif

#define NB 8
#define ND 6
#define NCH 48
#define HH 256
#define WW 256
#define HWp (HH*WW)
#define H2 128
#define W2 128
#define HW2 (H2*W2)
#define H1D 258
#define H3D 130

// float offsets inside the stats block S (first 4 KB of ws)
#define S_XS 0
#define S_XP 8
#define S_CE_SC 48
#define S_CE_BI 96
#define S_MD_SUM 144
#define S_MD_SQ 192
#define S_MD_SC 240
#define S_MD_BI 288
#define S_CM_SUM 336
#define S_CM_SQ 384
#define S_CM_SC 432
#define S_CM_BI 480
#define S_BO_SUM 528
#define S_BO_SQ 576
#define S_BO_SC 624
#define S_BO_BI 672
#define S_NFLOATS 720

__constant__ int TPX[8] = {0,1,1,1,0,-1,-1,-1};
__constant__ int TPY[8] = {1,1,0,-1,-1,-1,0,1};

__device__ inline float sigmoidf_(float v) { return 1.f / (1.f + __expf(-v)); }

// LDS byte offset of a shared-memory object: generic AMDGPU LDS addresses are
// aperture_hi | offset, so the low 32 bits are the LDS byte address.
__device__ inline unsigned lds_off(const void* p) {
  return (unsigned)(unsigned long long)p;
}

#ifdef USE_TDM
// Tensor Data Mover: 1-D copy of ndw dwords from global to LDS.
// D# per CDNA5 ISA: group0 = {count, lds_addr, global_addr, type=2},
// group1 = {data_size=4B, tensor_dim0/1, tile_dim0/1, tensor_dim0_stride}.
__device__ inline void tdm_load_1d(const void* gsrc, unsigned ldsoff, unsigned ndw) {
  unsigned long long ga = (unsigned long long)gsrc;
  u32x4 g0 = { 1u,                                   // count=1 (valid descriptor)
               ldsoff,                               // lds_addr (bytes)
               (unsigned)ga,                         // global_addr[31:0]
               (unsigned)(((ga >> 32) & 0x1FFFFFFull) | (2u << 30)) }; // [56:32] | type=2
  i32x8 g1 = { (int)(2u << 16),                      // data_size = 2 (4 bytes)
               (int)((ndw & 0xFFFFu) << 16),         // tensor_dim0[15:0]  -> bits 63:48
               (int)(((ndw >> 16) & 0xFFFFu) | (1u << 16)), // dim0[31:16] | tensor_dim1=1
               (int)((ndw & 0xFFFFu) << 16),         // tile_dim0 -> bits 127:112
               1,                                    // tile_dim1 = 1
               (int)ndw,                             // tensor_dim0_stride[31:0]
               0, 0 };
  i32x4 z4 = { 0, 0, 0, 0 };
#if __clang_major__ >= 23
  i32x8 z8 = { 0, 0, 0, 0, 0, 0, 0, 0 };
  __builtin_amdgcn_tensor_load_to_lds(g0, g1, z4, z4, z8, 0);
#else
  __builtin_amdgcn_tensor_load_to_lds(g0, g1, z4, z4, 0);
#endif
}
#endif

// ---------------- stats / params ----------------

__global__ void k_zero(float* S) {
  for (int i = threadIdx.x; i < S_NFLOATS; i += blockDim.x) S[i] = 0.f;
}

__global__ void k_xstats(const float* __restrict__ x, float* __restrict__ S) {
  __shared__ float ls[48];
  int tid = threadIdx.x;
  if (tid < 48) ls[tid] = 0.f;
  __syncthreads();
  long p = (long)blockIdx.x * blockDim.x + tid;
  if (p < (long)NB * HWp) {
    long b = p / HWp, pix = p % HWp;
    float v[ND];
    #pragma unroll
    for (int d = 0; d < ND; ++d) v[d] = x[((long)(b * ND + d)) * HWp + pix];
    #pragma unroll
    for (int d = 0; d < ND; ++d) {
      atomicAdd(&ls[S_XS + d], v[d]);
      #pragma unroll
      for (int e = 0; e < ND; ++e) atomicAdd(&ls[S_XP + d * ND + e], v[d] * v[e]);
    }
  }
  __syncthreads();
  if (tid < 44 && ls[tid] != 0.f) atomicAdd(&S[tid], ls[tid]);
}

__global__ void k_ce_params(const float* __restrict__ w1, const float* __restrict__ g,
                            const float* __restrict__ b, float* __restrict__ S) {
  int c = threadIdx.x;
  if (c >= NCH) return;
  const float inv = 1.f / (float)((long)NB * HWp);
  float m[ND], C[ND * ND];
  for (int d = 0; d < ND; ++d) m[d] = S[S_XS + d] * inv;
  for (int d = 0; d < ND; ++d)
    for (int e = 0; e < ND; ++e) C[d * ND + e] = S[S_XP + d * ND + e] * inv - m[d] * m[e];
  float mu = 0.f, var = 0.f;
  for (int d = 0; d < ND; ++d) mu += w1[c * ND + d] * m[d];
  for (int d = 0; d < ND; ++d)
    for (int e = 0; e < ND; ++e) var += w1[c * ND + d] * w1[c * ND + e] * C[d * ND + e];
  float sc = g[c] * rsqrtf(var + 1e-5f);
  S[S_CE_SC + c] = sc;
  S[S_CE_BI + c] = b[c] - mu * sc;
}

__global__ void k_y(const float* __restrict__ x, const float* __restrict__ w1,
                    const float* __restrict__ w2, const float* __restrict__ S,
                    float* __restrict__ ybuf) {
  long p = (long)blockIdx.x * blockDim.x + threadIdx.x;
  if (p >= (long)NB * HWp) return;
  long b = p / HWp, pix = p % HWp;
  float v[ND];
  #pragma unroll
  for (int d = 0; d < ND; ++d) v[d] = x[((long)(b * ND + d)) * HWp + pix];
  float acc = 0.f;
  for (int c = 0; c < NCH; ++c) {
    float a = 0.f;
    #pragma unroll
    for (int d = 0; d < ND; ++d) a += w1[c * ND + d] * v[d];
    acc += w2[c] * tanhf(a * S[S_CE_SC + c] + S[S_CE_BI + c]);
  }
  ybuf[p] = acc;
}

// ---------------- merge-determinant front conv (pad-2, 10ch -> 48ch) ----------------
// writes h1 CHANNEL-LAST: h1[b][row][col][48]; vectorized 16B stores

__global__ __launch_bounds__(128) void k_md_ct(const float* __restrict__ x,
                                               const float* __restrict__ ybuf,
                                               const float* __restrict__ w,
                                               _Float16* __restrict__ h1) {
  __shared__ float lw[NCH * 10 * 9];
  int tid = threadIdx.x;
  for (int i = tid; i < NCH * 10 * 9; i += 128) lw[i] = w[i];
  __syncthreads();
  long t = (long)blockIdx.x * 128 + tid;
  if (t >= (long)NB * H1D * H1D) return;
  int b = (int)(t / (H1D * H1D));
  int rem = (int)(t % (H1D * H1D));
  int yy = rem / H1D, xx = rem % H1D;
  float acc[NCH];
  #pragma unroll
  for (int o = 0; o < NCH; ++o) acc[o] = 0.f;
  for (int dy = 0; dy < 3; ++dy)
    for (int dx = 0; dx < 3; ++dx) {
      int iy = yy + dy - 2, ix = xx + dx - 2;
      bool inb = (iy >= 0) & (iy < HH) & (ix >= 0) & (ix < WW);
      int tap = dy * 3 + dx;
      for (int cch = 0; cch < 10; ++cch) {
        float tev = 0.f;
        if (inb) {
          if (cch < 2) {
            tev = x[((long)(b * ND + cch)) * HWp + (long)iy * WW + ix];
          } else {
            int k = cch - 2;
            int sy = (iy + TPY[k] + HH) & (HH - 1);
            int sx = (ix - TPX[k] + WW) & (WW - 1);
            tev = ybuf[(long)b * HWp + (long)sy * WW + sx];
          }
        }
        #pragma unroll
        for (int o = 0; o < NCH; ++o) acc[o] += lw[(o * 10 + cch) * 9 + tap] * tev;
      }
    }
  long base = ((long)b * (H1D * H1D) + rem) * NCH;
  union PK8 { uint4 q; _Float16 e[8]; };
  #pragma unroll
  for (int ck = 0; ck < 6; ++ck) {
    PK8 pk;
    #pragma unroll
    for (int j = 0; j < 8; ++j) pk.e[j] = (_Float16)tanhf(acc[ck * 8 + j]);
    ((uint4*)(h1 + base))[ck] = pk.q;
  }
}

// ---------------- pack weights into per-lane WMMA A fragments ----------------
// A 16x32 f16 packing used here (contiguous k): lane<16 -> M=lane, halves 0..15 = K0..15
//                    lane>=16 -> M=lane-16, halves 0..15 = K16..31

__global__ void k_pack(const float* __restrict__ w, _Float16* __restrict__ wp) {
  int t = blockIdx.x * blockDim.x + threadIdx.x;
  if (t >= 3 * 14 * 32) return;
  int lane = t & 31;
  int kc = (t >> 5) % 14;
  int mt = (t >> 5) / 14;
  int m = mt * 16 + (lane & 15);
  int khi = (lane < 16) ? 0 : 16;
  for (int h = 0; h < 16; ++h) {
    int k = kc * 32 + khi + h;
    float v = 0.f;
    if (k < 432) {
      int tap = k / 48, ch = k - tap * 48;
      v = w[((long)m * 48 + ch) * 9 + tap];
    }
    wp[(long)t * 16 + h] = (_Float16)v;
  }
}

__global__ void k_pack_cm(const float* __restrict__ w, _Float16* __restrict__ wp) {
  int t = blockIdx.x * blockDim.x + threadIdx.x;
  if (t >= 3 * 5 * 32) return;
  int lane = t & 31;
  int kc = (t >> 5) % 5;
  int g = (t >> 5) / 5;
  int m = lane & 15;
  int khi = (lane < 16) ? 0 : 16;
  for (int h = 0; h < 16; ++h) {
    int k = kc * 32 + khi + h;
    float v = 0.f;
    if (k < 144) {
      int tap = k >> 4, ic = k & 15;
      v = w[(long)((g * 16 + m) * 16 + ic) * 9 + tap];
    }
    wp[(long)t * 16 + h] = (_Float16)v;
  }
}

// ---------------- heavy layer: implicit-GEMM 3x3 conv 48->48 with WMMA ----------------
// Weight pack (43 KB) staged into LDS via the Tensor Data Mover; hot loop is
// pure LDS (ds_load_b128 fragments) + v_wmma. 84 WMMAs per wave.

__global__ __launch_bounds__(128) void k_md_c1(const _Float16* __restrict__ h1,
                                               const _Float16* __restrict__ wp,
                                               _Float16* __restrict__ cout,
                                               float* __restrict__ Sg) {
  __shared__ __align__(16) _Float16 patch[10][18][56];
  __shared__ __align__(16) _Float16 wlds[3 * 14 * 32 * 16];  // 43008 B
  __shared__ float lsum[NCH], lsq[NCH];
  int tid = threadIdx.x;
  int b = blockIdx.z;
  int y0 = blockIdx.y * 8;
  int x0 = blockIdx.x * 16;
  if (tid < NCH) { lsum[tid] = 0.f; lsq[tid] = 0.f; }

#ifdef USE_TDM
  if (tid < 32) tdm_load_1d(wp, lds_off(wlds), 3 * 14 * 32 * 16 / 2);  // 10752 dwords
#else
  {
    const uint4* gs = (const uint4*)wp;
    uint4* dst = (uint4*)wlds;
    for (int i = tid; i < 3 * 14 * 32 * 2; i += 128) dst[i] = gs[i];
  }
#endif

  const long h1s = (long)H1D * H1D;
  for (int i = tid; i < 10 * 18 * 6; i += 128) {
    int ck = i % 6;
    int cl = (i / 6) % 18;
    int r = i / (6 * 18);
    const uint4* src = (const uint4*)(h1 + ((long)b * h1s + (long)(y0 + r) * H1D + (x0 + cl)) * NCH);
    *(uint4*)&patch[r][cl][ck * 8] = src[ck];
  }
#ifdef USE_TDM
  __builtin_amdgcn_s_wait_tensorcnt(0);
#endif
  __syncthreads();

  int w = tid >> 5;
  int lane = tid & 31;
  int n = lane & 15;                 // B-matrix column = output col
  int ahalf = (lane < 16) ? 0 : 1;   // B k order used here (matches A pack): lanes 0-15 K0..15, lanes 16-31 K16..31
  v8f acc[2][3] = {};

  for (int kc = 0; kc < 14; ++kc) {
    int a = kc * 2 + ahalf;          // this lane's 16-aligned K-run
    V16 bf0, bf1;
    if (a < 27) {
      int tap = a / 3;
      int chb = (a - tap * 3) * 16;
      int dy = tap / 3, dx = tap - dy * 3;
      const _Float16* s0 = &patch[w + dy][n + dx][chb];
      const _Float16* s1 = &patch[w + 4 + dy][n + dx][chb];
      ((uint4*)bf0.e)[0] = *(const uint4*)s0;
      ((uint4*)bf0.e)[1] = *(const uint4*)(s0 + 8);
      ((uint4*)bf1.e)[0] = *(const uint4*)s1;
      ((uint4*)bf1.e)[1] = *(const uint4*)(s1 + 8);
    } else {
      #pragma unroll
      for (int h = 0; h < 16; ++h) { bf0.e[h] = (_Float16)0.f; bf1.e[h] = (_Float16)0.f; }
    }
    #pragma unroll
    for (int mt = 0; mt < 3; ++mt) {
      v16h am = *(const v16h*)&wlds[((mt * 14 + kc) * 32 + lane) * 16];
      acc[0][mt] = __builtin_amdgcn_wmma_f32_16x16x32_f16(false, am, false, bf0.v, (short)0, acc[0][mt], false, false);
      acc[1][mt] = __builtin_amdgcn_wmma_f32_16x16x32_f16(false, am, false, bf1.v, (short)0, acc[1][mt], false, false);
    }
  }

  // C/D layout: VGPR r holds M = r + (lane<16 ? 0 : 8), N = lane%16
  const long cs = (long)HWp;
  int ox = x0 + n;
  #pragma unroll
  for (int half = 0; half < 2; ++half) {
    int oy = y0 + w + half * 4;
    long pix = (long)oy * WW + ox;
    #pragma unroll
    for (int r8 = 0; r8 < 8; ++r8) {
      int mrow = r8 + ((lane < 16) ? 0 : 8);
      #pragma unroll
      for (int mt = 0; mt < 3; ++mt) {
        float v = tanhf(acc[half][mt][r8]);
        cout[((long)(b * NCH + mt * 16 + mrow)) * cs + pix] = (_Float16)v;
        atomicAdd(&lsum[mt * 16 + mrow], v);
        atomicAdd(&lsq[mt * 16 + mrow], v * v);
      }
    }
  }
  __syncthreads();
  if (tid < NCH) {
    atomicAdd(&Sg[S_MD_SUM + tid], lsum[tid]);
    atomicAdd(&Sg[S_MD_SQ + tid], lsq[tid]);
  }
}

// ---------------- generic BN finalize ----------------

__global__ void k_bn_fin(float* __restrict__ S, int so, int qo, int co, int bo,
                         const float* __restrict__ g, const float* __restrict__ b,
                         float count) {
  int c = threadIdx.x;
  if (c >= NCH) return;
  float m = S[so + c] / count;
  float v = S[qo + c] / count - m * m;
  float sc = g[c] * rsqrtf(v + 1e-5f);
  S[co + c] = sc;
  S[bo + c] = b[c] - m * sc;
}

// ---------------- CHANGE = sigmoid(conv2x2s2(tanh(affine(c)))) ----------------

__global__ void k_change(const _Float16* __restrict__ cbuf, const float* __restrict__ w2,
                         const float* __restrict__ S, float* __restrict__ CH) {
  long p = (long)blockIdx.x * blockDim.x + threadIdx.x;
  if (p >= (long)NB * HW2) return;
  long b = p / HW2;
  int q = (int)(p % HW2);
  int yy = q / W2, xx = q % W2;
  float acc = 0.f;
  for (int ch = 0; ch < NCH; ++ch) {
    float sc = S[S_MD_SC + ch], bi = S[S_MD_BI + ch];
    #pragma unroll
    for (int ky = 0; ky < 2; ++ky)
      #pragma unroll
      for (int kx = 0; kx < 2; ++kx) {
        float v = (float)cbuf[((long)(b * NCH + ch)) * HWp + (long)(2 * yy + ky) * WW + (2 * xx + kx)];
        acc += w2[ch * 4 + ky * 2 + kx] * tanhf(v * sc + bi);
      }
  }
  CH[p] = sigmoidf_(acc);
}

// ---------------- cell-merge front: 2x2 s2 grouped conv + BN stats ----------------

__global__ void k_cm1(const float* __restrict__ x, const float* __restrict__ w,
                      _Float16* __restrict__ raw, float* __restrict__ S) {
  __shared__ float s1, s2;
  int ch = blockIdx.y, b = blockIdx.z;
  if (threadIdx.x == 0) { s1 = 0.f; s2 = 0.f; }
  __syncthreads();
  int q = blockIdx.x * blockDim.x + threadIdx.x;
  int yy = q / W2, xx = q % W2;
  int g = ch / 16;
  float acc = 0.f;
  #pragma unroll
  for (int ic = 0; ic < 2; ++ic)
    #pragma unroll
    for (int ky = 0; ky < 2; ++ky)
      #pragma unroll
      for (int kx = 0; kx < 2; ++kx)
        acc += x[((long)(b * ND + g * 2 + ic)) * HWp + (long)(2 * yy + ky) * WW + (2 * xx + kx)] *
               w[((ch * 2 + ic) * 2 + ky) * 2 + kx];
  raw[((long)(b * NCH + ch)) * HW2 + q] = (_Float16)acc;
  atomicAdd(&s1, acc);
  atomicAdd(&s2, acc * acc);
  __syncthreads();
  if (threadIdx.x == 0) {
    atomicAdd(&S[S_CM_SUM + ch], s1);
    atomicAdd(&S[S_CM_SQ + ch], s2);
  }
}

// ---------------- cell-merge funnel: grouped 3x3 pad2 conv as WMMA (M=16/group) -------

__global__ __launch_bounds__(128) void k_cm_ct(const _Float16* __restrict__ raw,
                                               const _Float16* __restrict__ wp,
                                               const float* __restrict__ S,
                                               _Float16* __restrict__ h3) {
  __shared__ __align__(16) _Float16 patch[6][18][24];
  __shared__ __align__(16) _Float16 wlds2[5 * 32 * 16];  // this group's 5120 B
  int tid = threadIdx.x;
  int b = blockIdx.z, g = blockIdx.y;
  int ty = blockIdx.x / 9, tx = blockIdx.x % 9;  // 33 y-tiles x 9 x-tiles
  int y0 = ty * 4, x0 = tx * 16;

#ifdef USE_TDM
  if (tid < 32) tdm_load_1d(wp + (long)g * (5 * 32 * 16), lds_off(wlds2), 5 * 32 * 16 / 2);
#else
  {
    const uint4* gs = (const uint4*)(wp + (long)g * (5 * 32 * 16));
    uint4* dst = (uint4*)wlds2;
    for (int i = tid; i < 5 * 32 * 2; i += 128) dst[i] = gs[i];
  }
#endif

  for (int i = tid; i < 6 * 18 * 16; i += 128) {
    int cl = i % 18;
    int ic = (i / 18) & 15;
    int r = i / (18 * 16);
    int iy = y0 - 2 + r, ix = x0 - 2 + cl;
    float v = 0.f;
    if (iy >= 0 && iy < H2 && ix >= 0 && ix < W2) {
      float rv = (float)raw[((long)(b * NCH + g * 16 + ic)) * HW2 + (long)iy * W2 + ix];
      v = tanhf(rv * S[S_CM_SC + g * 16 + ic] + S[S_CM_BI + g * 16 + ic]);
    }
    patch[r][cl][ic] = (_Float16)v;
  }
#ifdef USE_TDM
  __builtin_amdgcn_s_wait_tensorcnt(0);
#endif
  __syncthreads();

  int w = tid >> 5, lane = tid & 31, n = lane & 15;
  int ahalf = (lane < 16) ? 0 : 1;
  v8f acc = {};
  for (int kc = 0; kc < 5; ++kc) {
    int a = kc * 2 + ahalf;          // tap index; a==9 is the zero-pad run
    V16 bf;
    if (a < 9) {
      int dy = a / 3, dx = a - dy * 3;
      const _Float16* s0 = &patch[w + dy][n + dx][0];
      ((uint4*)bf.e)[0] = *(const uint4*)s0;
      ((uint4*)bf.e)[1] = *(const uint4*)(s0 + 8);
    } else {
      #pragma unroll
      for (int h = 0; h < 16; ++h) bf.e[h] = (_Float16)0.f;
    }
    v16h am = *(const v16h*)&wlds2[(kc * 32 + lane) * 16];
    acc = __builtin_amdgcn_wmma_f32_16x16x32_f16(false, am, false, bf.v, (short)0, acc, false, false);
  }
  int oy = y0 + w, ox = x0 + n;
  if (oy < H3D && ox < H3D) {
    long hs = (long)H3D * H3D;
    #pragma unroll
    for (int r8 = 0; r8 < 8; ++r8) {
      int m = r8 + ((lane < 16) ? 0 : 8);
      h3[((long)(b * NCH + g * 16 + m)) * hs + (long)oy * H3D + ox] = (_Float16)tanhf(acc[r8]);
    }
  }
}

__global__ void k_cm_c2(const _Float16* __restrict__ h3, const float* __restrict__ w,
                        float* __restrict__ xm) {
  long t = (long)blockIdx.x * blockDim.x + threadIdx.x;
  if (t >= (long)NB * ND * HW2) return;
  long b = t / ((long)ND * HW2);
  int rem = (int)(t % ((long)ND * HW2));
  int d = rem / HW2, q = rem % HW2;
  int yy = q / W2, xx = q % W2;
  int g = d / 2;
  const long hs = (long)H3D * H3D;
  float acc = 0.f;
  for (int ic = 0; ic < 16; ++ic)
    #pragma unroll
    for (int ky = 0; ky < 3; ++ky)
      #pragma unroll
      for (int kx = 0; kx < 3; ++kx)
        acc += (float)h3[((long)(b * NCH + g * 16 + ic)) * hs + (long)(yy + ky) * H3D + (xx + kx)] *
               w[((d * 16 + ic) * 3 + ky) * 3 + kx];
  xm[t] = tanhf(acc);
}

// ---------------- born 1x1 front conv + BN stats (channel-last raw) ----------------

__global__ void k_bo1(const float* __restrict__ xm, const float* __restrict__ w,
                      _Float16* __restrict__ raw, float* __restrict__ S) {
  __shared__ float lsum[NCH], lsq[NCH];
  int tid = threadIdx.x;
  if (tid < NCH) { lsum[tid] = 0.f; lsq[tid] = 0.f; }
  __syncthreads();
  long p = (long)blockIdx.x * blockDim.x + tid;
  long b = p / HW2;
  int q = (int)(p % HW2);
  float v[ND];
  #pragma unroll
  for (int d = 0; d < ND; ++d) v[d] = xm[((long)(b * ND + d)) * HW2 + q];
  union PK { uint4 u[6]; _Float16 e[48]; } pk;
  for (int ch = 0; ch < NCH; ++ch) {
    float a = 0.f;
    #pragma unroll
    for (int d = 0; d < ND; ++d) a += v[d] * w[ch * ND + d];
    pk.e[ch] = (_Float16)a;
    atomicAdd(&lsum[ch], a);
    atomicAdd(&lsq[ch], a * a);
  }
  uint4* dst = (uint4*)(raw + p * NCH);
  #pragma unroll
  for (int j = 0; j < 6; ++j) dst[j] = pk.u[j];
  __syncthreads();
  if (tid < NCH) {
    atomicAdd(&S[S_BO_SUM + tid], lsum[tid]);
    atomicAdd(&S[S_BO_SQ + tid], lsq[tid]);
  }
}

// ---------------- born tail + blend + output sigmoid ----------------

__global__ void k_final(const float* __restrict__ xm, const _Float16* __restrict__ raw,
                        const float* __restrict__ w2, const float* __restrict__ CH,
                        const float* __restrict__ S, float* __restrict__ out) {
  long p = (long)blockIdx.x * blockDim.x + threadIdx.x;
  if (p >= (long)NB * HW2) return;
  long b = p / HW2;
  int q = (int)(p % HW2);
  const _Float16* rp = raw + p * NCH;  // channel-last: 96 contiguous bytes
  float t48[NCH];
  for (int ch = 0; ch < NCH; ++ch)
    t48[ch] = tanhf((float)rp[ch] * S[S_BO_SC + ch] + S[S_BO_BI + ch]);
  float cv = CH[p];
  for (int d = 0; d < ND; ++d) {
    float born = 0.f;
    for (int ch = 0; ch < NCH; ++ch) born += w2[d * NCH + ch] * t48[ch];
    float xv = xm[((long)(b * ND + d)) * HW2 + q];
    float o = xv * (1.f - cv) + cv * born;
    if (d < 3) o = sigmoidf_(o);
    out[((long)(b * ND + d)) * HW2 + q] = o;
  }
}

// ---------------- host-side orchestration ----------------

extern "C" void kernel_launch(void* const* d_in, const int* in_sizes, int n_in,
                              void* d_out, int out_size, void* d_ws, size_t ws_size,
                              hipStream_t stream) {
  const float* x     = (const float*)d_in[0];
  const float* ce_w1 = (const float*)d_in[1];
  const float* ce_g  = (const float*)d_in[2];
  const float* ce_b  = (const float*)d_in[3];
  const float* ce_w2 = (const float*)d_in[4];
  const float* md_ct = (const float*)d_in[5];
  const float* md_c1 = (const float*)d_in[6];
  const float* md_g  = (const float*)d_in[7];
  const float* md_b  = (const float*)d_in[8];
  const float* md_c2 = (const float*)d_in[9];
  const float* cm_w1 = (const float*)d_in[10];
  const float* cm_g  = (const float*)d_in[11];
  const float* cm_b  = (const float*)d_in[12];
  const float* cm_ct = (const float*)d_in[13];
  const float* cm_c2 = (const float*)d_in[14];
  const float* bo_w1 = (const float*)d_in[15];
  const float* bo_g  = (const float*)d_in[16];
  const float* bo_b  = (const float*)d_in[17];
  const float* bo_w2 = (const float*)d_in[18];
  float* out = (float*)d_out;
  (void)in_sizes; (void)n_in; (void)out_size; (void)ws_size;

  char* ws = (char*)d_ws;
  size_t off = 0;
  float* S = (float*)(ws + off);              off += 4096;
  float* ybuf = (float*)(ws + off);           off += (size_t)NB * HWp * 4;
  _Float16* wpack = (_Float16*)(ws + off);    off += 65536;
  _Float16* wpack2 = (_Float16*)(ws + off);   off += 16384;
  size_t h1base = off;
  _Float16* h1 = (_Float16*)(ws + off);       off += (size_t)NB * H1D * H1D * NCH * 2;
  _Float16* cbuf = (_Float16*)(ws + off);     off += (size_t)NB * NCH * HWp * 2;
  float* CHb = (float*)(ws + off);            off += (size_t)NB * HW2 * 4;
  // cell-merge / born path overlays the dead h1 region
  size_t r = h1base;
  _Float16* cm1 = (_Float16*)(ws + r);        r += (size_t)NB * NCH * HW2 * 2;
  _Float16* h3  = (_Float16*)(ws + r);        r += (size_t)NB * NCH * H3D * H3D * 2;
  float* xm = (float*)(ws + r);               r += (size_t)NB * ND * HW2 * 4;
  _Float16* bo1 = (_Float16*)(ws + r);

  const long P = (long)NB * HWp;     // 524288
  const long P2 = (long)NB * HW2;    // 131072

  k_zero<<<1, 256, 0, stream>>>(S);
  k_xstats<<<(int)((P + 255) / 256), 256, 0, stream>>>(x, S);
  k_ce_params<<<1, 64, 0, stream>>>(ce_w1, ce_g, ce_b, S);
  k_y<<<(int)((P + 255) / 256), 256, 0, stream>>>(x, ce_w1, ce_w2, S, ybuf);

  long mdct_tot = (long)NB * H1D * H1D;
  k_md_ct<<<(int)((mdct_tot + 127) / 128), 128, 0, stream>>>(x, ybuf, md_ct, h1);

  k_pack<<<(3 * 14 * 32 + 63) / 64, 64, 0, stream>>>(md_c1, wpack);
  k_pack_cm<<<(3 * 5 * 32 + 63) / 64, 64, 0, stream>>>(cm_ct, wpack2);

  k_md_c1<<<dim3(WW / 16, HH / 8, NB), 128, 0, stream>>>(h1, wpack, cbuf, S);
  k_bn_fin<<<1, 64, 0, stream>>>(S, S_MD_SUM, S_MD_SQ, S_MD_SC, S_MD_BI, md_g, md_b, (float)P);
  k_change<<<(int)((P2 + 255) / 256), 256, 0, stream>>>(cbuf, md_c2, S, CHb);

  k_cm1<<<dim3(HW2 / 256, NCH, NB), 256, 0, stream>>>(x, cm_w1, cm1, S);
  k_bn_fin<<<1, 64, 0, stream>>>(S, S_CM_SUM, S_CM_SQ, S_CM_SC, S_CM_BI, cm_g, cm_b, (float)P2);
  k_cm_ct<<<dim3(33 * 9, 3, NB), 128, 0, stream>>>(cm1, wpack2, S, h3);
  long c2_tot = (long)NB * ND * HW2;
  k_cm_c2<<<(int)((c2_tot + 255) / 256), 256, 0, stream>>>(h3, cm_c2, xm);

  k_bo1<<<(int)(P2 / 256), 256, 0, stream>>>(xm, bo_w1, bo1, S);
  k_bn_fin<<<1, 64, 0, stream>>>(S, S_BO_SUM, S_BO_SQ, S_BO_SC, S_BO_BI, bo_g, bo_b, (float)P2);
  k_final<<<(int)((P2 + 255) / 256), 256, 0, stream>>>(xm, bo1, bo_w2, CHb, S, out);
}
